// MultiHeadAttention_61263413510778
// MI455X (gfx1250) — matrix operand order repaired and checked by reading, not compiled, not run on hardware
//
#include <hip/hip_runtime.h>
#include <hip/hip_bf16.h>
#include <cstdint>

#define B_ 2
#define S_ 2048
#define D_ 2048
#define H_ 16
#define HD_ 128
#define M_TOTAL (B_*S_)   // 4096

#define USE_ASYNC_LDS 1

typedef __attribute__((ext_vector_type(16))) _Float16 v16h;
typedef __attribute__((ext_vector_type(8)))  _Float16 v8h;
typedef __attribute__((ext_vector_type(8)))  float    v8f;

static __device__ __forceinline__ v8f zero_v8f() {
  v8f z;
#pragma unroll
  for (int i = 0; i < 8; ++i) z[i] = 0.0f;
  return z;
}

// A fragment: 16x32 f16 (MxK). Lanes 0-15: M=lane, K=0..15;
// lanes 16-31: M=lane-16, K=16..31.
static __device__ __forceinline__ v16h load_frag_a(const _Float16* base, int ld) {
  const int lane = threadIdx.x & 31;
  const int m  = lane & 15;
  const int k0 = (lane >> 4) << 4;     // 0 or 16
  const _Float16* p = base + m * ld + k0;
  v16h r;
#pragma unroll
  for (int i = 0; i < 16; ++i) r[i] = p[i];
  return r;
}

// B fragment: 32x16 f16 (KxN), B[k][n] = src[n*ld + k] (contiguous over K per lane).
static __device__ __forceinline__ v16h load_frag_b(const _Float16* base, int ld) {
  const int lane = threadIdx.x & 31;
  const int n  = lane & 15;
  const int k0 = (lane >> 4) << 4;     // 0 or 16
  const _Float16* p = base + n * ld + k0;
  v16h r;
#pragma unroll
  for (int i = 0; i < 16; ++i) r[i] = p[i];
  return r;
}

// Async copy of one 16-byte chunk per lane: global -> LDS (ASYNCcnt-tracked).
static __device__ __forceinline__ void async_b128(const _Float16* gptr, _Float16* lptr) {
#if USE_ASYNC_LDS
  uint32_t lds_off = (uint32_t)(uintptr_t)lptr;       // LDS aperture keeps offset in addr[31:0]
  asm volatile("global_load_async_to_lds_b128 %0, %1, off"
               :: "v"(lds_off), "v"((uint64_t)(uintptr_t)gptr) : "memory");
#else
  *(v8h*)lptr = *(const v8h*)gptr;
#endif
}

static __device__ __forceinline__ void wait_async(int n) {
#if USE_ASYNC_LDS
  if (n == 0) asm volatile("s_wait_asynccnt 0" ::: "memory");
  else        asm volatile("s_wait_asynccnt 4" ::: "memory");
#else
  (void)n;
#endif
}

// ---------------------------------------------------------------- convert
__global__ void k_f32_to_f16(const float* __restrict__ in, _Float16* __restrict__ out, int n) {
  int i = blockIdx.x * blockDim.x + threadIdx.x;
  int stride = gridDim.x * blockDim.x;
  for (; i < n; i += stride) out[i] = (_Float16)in[i];
}

// ---------------------------------------------------------------- shared GEMM stage loader
// A tile: 128x32 halfs, B tile: 128x32 halfs. 256 threads, 2 reps each per tile.
static __device__ __forceinline__ void stage_tiles(const _Float16* __restrict__ Xg,
                                                   const _Float16* __restrict__ Wg,
                                                   _Float16* sAbuf, _Float16* sBbuf,
                                                   int m0, int n0, int k0, int tid) {
#pragma unroll
  for (int rep = 0; rep < 2; ++rep) {
    int idx = rep * 256 + tid;         // 0..511
    int row = idx >> 2;                // 0..127
    int col = (idx & 3) * 8;           // 0,8,16,24
    async_b128(&Xg[(size_t)(m0 + row) * D_ + k0 + col], &sAbuf[row * 32 + col]);
    async_b128(&Wg[(size_t)(n0 + row) * D_ + k0 + col], &sBbuf[row * 32 + col]);
  }
}

// ---------------------------------------------------------------- QKV proj + RoPE
// Y[m,e] = sum_k Xh[m,k]*Wh[e,k]. Block tile 128x128, BK=32, double-buffered async LDS.
// 8 waves as 4(M)x2(N); each wave: 32x64 = 2x4 WMMA tiles.
__global__ __launch_bounds__(256)
void k_qkv_proj(const _Float16* __restrict__ Xh, const _Float16* __restrict__ Wh,
                _Float16* __restrict__ Out, const float* __restrict__ fcos,
                const float* __restrict__ fsin, int apply_rope)
{
  __shared__ _Float16 sA[2][128 * 32];
  __shared__ _Float16 sB[2][128 * 32];

  const int tid = threadIdx.x;
  const int wid = tid >> 5;
  const int m0  = blockIdx.x * 128;
  const int n0  = blockIdx.y * 128;
  const int wm  = wid & 3;             // 0..3 -> 32-row slab
  const int wn  = wid >> 2;            // 0..1 -> 64-col slab
  const int lane = tid & 31;
  const int lo = lane & 15, hi = lane >> 4;

  v8f acc[2][4];
#pragma unroll
  for (int i = 0; i < 2; ++i)
#pragma unroll
    for (int j = 0; j < 4; ++j) acc[i][j] = zero_v8f();

  const int NK = D_ / 32;
  stage_tiles(Xh, Wh, sA[0], sB[0], m0, n0, 0, tid);

  for (int ks = 0; ks < NK; ++ks) {
    const int buf = ks & 1;
    if (ks + 1 < NK) {
      stage_tiles(Xh, Wh, sA[buf ^ 1], sB[buf ^ 1], m0, n0, (ks + 1) * 32, tid);
      wait_async(4);
    } else {
      wait_async(0);
    }
    __syncthreads();

    v16h af[2], bf[4];
#pragma unroll
    for (int i = 0; i < 2; ++i) af[i] = load_frag_a(&sA[buf][(wm * 32 + i * 16) * 32], 32);
#pragma unroll
    for (int j = 0; j < 4; ++j) bf[j] = load_frag_b(&sB[buf][(wn * 64 + j * 16) * 32], 32);
#pragma unroll
    for (int i = 0; i < 2; ++i)
#pragma unroll
      for (int j = 0; j < 4; ++j)
        acc[i][j] = __builtin_amdgcn_wmma_f32_16x16x32_f16(false, af[i], false, bf[j],
                                                           (short)0, acc[i][j], false, false);
    __syncthreads();
  }

  // Epilogue: RoPE via lane-pair shuffle (pair (hd, hd^1) sits in adjacent lanes).
#pragma unroll
  for (int i = 0; i < 2; ++i) {
#pragma unroll
    for (int j = 0; j < 4; ++j) {
      const int e = n0 + wn * 64 + j * 16 + lo;
      const int h  = e >> 7;
      const int hd = e & (HD_ - 1);
      const int fi = hd >> 1;
#pragma unroll
      for (int r = 0; r < 8; ++r) {
        int gm = m0 + wm * 32 + i * 16 + r + 8 * hi;
        int bb = gm >> 11;             // / S_
        int s  = gm & (S_ - 1);
        float v = acc[i][j][r];
        float o = v;
        if (apply_rope) {
          float w  = __shfl_xor(v, 1, 32);
          float re = (hd & 1) ? w : v;
          float im = (hd & 1) ? v : w;
          float c  = fcos[s * (HD_ / 2) + fi];
          float sn = fsin[s * (HD_ / 2) + fi];
          o = (hd & 1) ? (re * sn + im * c) : (re * c - im * sn);
        }
        Out[(((size_t)(bb * H_ + h) * S_) + s) * HD_ + hd] = (_Float16)o;
      }
    }
  }
}

// ---------------------------------------------------------------- causal flash attention
__global__ __launch_bounds__(256)
void k_attention(const _Float16* __restrict__ Qh, const _Float16* __restrict__ Kh,
                 const _Float16* __restrict__ Vh, _Float16* __restrict__ Oh)
{
  __shared__ _Float16 sQ [32 * 128];
  __shared__ _Float16 sK [64 * 128];
  __shared__ _Float16 sVt[128 * 64];   // transposed: [hd][kv]
  __shared__ float    sS [32 * 64];
  __shared__ _Float16 sP [32 * 64];
  __shared__ float    salpha[32];
  __shared__ float    sl[32];

  const int tid = threadIdx.x;
  const int wid = tid >> 5;
  const int q0  = blockIdx.x * 32;
  const int bh  = blockIdx.y;
  const int b   = bh >> 4;
  const int h   = bh & 15;

  const _Float16* Qbase = Qh + (size_t)bh * S_ * HD_;
  const _Float16* Kbase = Kh + (size_t)bh * S_ * HD_;
  const _Float16* Vbase = Vh + (size_t)bh * S_ * HD_;

#pragma unroll
  for (int rep = 0; rep < 2; ++rep) {  // Q tile 32x128 via async LDS
    int idx = tid + rep * 256;
    int i   = idx >> 4;
    int hd0 = (idx & 15) * 8;
    async_b128(&Qbase[(q0 + i) * 128 + hd0], &sQ[i * 128 + hd0]);
  }

  const int mt   = wid >> 2;
  const int ntS  = wid & 3;
  const int lane = tid & 31;
  const int lo = lane & 15, hi = lane >> 4;
  const int nt0 = ntS * 2, nt1 = ntS * 2 + 1;

  const int srow = tid >> 3;
  const int st   = tid & 7;
  float m_run = -3.0e38f;
  float l_run = 0.0f;

  v8f accO0 = zero_v8f();
  v8f accO1 = zero_v8f();
  const float scale = 0.08838834764831845f;    // 1/sqrt(128)

  const int kv_steps = (q0 + 32 + 63) >> 6;
  for (int step = 0; step < kv_steps; ++step) {
    const int j0 = step * 64;
    __syncthreads();                   // prior readers of sK/sVt/sP done
#pragma unroll
    for (int rep = 0; rep < 4; ++rep) {
      int idx = tid + rep * 256;
      int kv  = idx >> 4;
      int hd0 = (idx & 15) * 8;
      async_b128(&Kbase[(j0 + kv) * 128 + hd0], &sK[kv * 128 + hd0]);
      v8h vv = *(const v8h*)&Vbase[(j0 + kv) * 128 + hd0];
#pragma unroll
      for (int x = 0; x < 8; ++x) sVt[(hd0 + x) * 64 + kv] = vv[x];
    }
    wait_async(0);
    __syncthreads();

    // scores: (32x64) = Q . K^T
    v8f sacc = zero_v8f();
#pragma unroll
    for (int kk = 0; kk < 128; kk += 32) {
      v16h a   = load_frag_a(&sQ[mt * 16 * 128 + kk], 128);
      v16h bfr = load_frag_b(&sK[ntS * 16 * 128 + kk], 128);
      sacc = __builtin_amdgcn_wmma_f32_16x16x32_f16(false, a, false, bfr, (short)0, sacc, false, false);
    }
    {
      int sk = j0 + ntS * 16 + lo;
#pragma unroll
      for (int r = 0; r < 8; ++r) {
        int sq = q0 + mt * 16 + r + 8 * hi;
        float v = sacc[r] * scale + (sk > sq ? -1.0e9f : 0.0f);
        sS[(mt * 16 + r + 8 * hi) * 64 + ntS * 16 + lo] = v;
      }
    }
    __syncthreads();

    { // online softmax, 8 threads per row
      float vals[8];
      float vmax = -3.0e38f;
#pragma unroll
      for (int x = 0; x < 8; ++x) {
        vals[x] = sS[srow * 64 + st * 8 + x];
        vmax = fmaxf(vmax, vals[x]);
      }
#pragma unroll
      for (int off = 1; off < 8; off <<= 1)
        vmax = fmaxf(vmax, __shfl_xor(vmax, off, 32));
      float mnew  = fmaxf(m_run, vmax);
      float alpha = __expf(m_run - mnew);
      float lsum  = 0.0f;
#pragma unroll
      for (int x = 0; x < 8; ++x) {
        float pv = __expf(vals[x] - mnew);
        sP[srow * 64 + st * 8 + x] = (_Float16)pv;
        lsum += pv;
      }
#pragma unroll
      for (int off = 1; off < 8; off <<= 1)
        lsum += __shfl_xor(lsum, off, 32);
      l_run = l_run * alpha + lsum;
      m_run = mnew;
      if (st == 0) salpha[srow] = alpha;
    }
    __syncthreads();

    { // O = alpha*O + P . V
      float ar[8];
#pragma unroll
      for (int r = 0; r < 8; ++r) ar[r] = salpha[mt * 16 + r + 8 * hi];
#pragma unroll
      for (int r = 0; r < 8; ++r) { accO0[r] *= ar[r]; accO1[r] *= ar[r]; }
#pragma unroll
      for (int kk = 0; kk < 64; kk += 32) {
        v16h a  = load_frag_a(&sP[mt * 16 * 64 + kk], 64);
        v16h b0 = load_frag_b(&sVt[nt0 * 16 * 64 + kk], 64);
        accO0 = __builtin_amdgcn_wmma_f32_16x16x32_f16(false, a, false, b0, (short)0, accO0, false, false);
        v16h b1 = load_frag_b(&sVt[nt1 * 16 * 64 + kk], 64);
        accO1 = __builtin_amdgcn_wmma_f32_16x16x32_f16(false, a, false, b1, (short)0, accO1, false, false);
      }
    }
  }

  __syncthreads();
  if (st == 0) sl[srow] = l_run;
  __syncthreads();
#pragma unroll
  for (int r = 0; r < 8; ++r) {
    float inv_l = 1.0f / sl[mt * 16 + r + 8 * hi];
    int s = q0 + mt * 16 + r + 8 * hi;
    Oh[((size_t)(b * S_ + s)) * D_ + h * HD_ + nt0 * 16 + lo] = (_Float16)(accO0[r] * inv_l);
    Oh[((size_t)(b * S_ + s)) * D_ + h * HD_ + nt1 * 16 + lo] = (_Float16)(accO1[r] * inv_l);
  }
}

// ---------------------------------------------------------------- output projection
__global__ __launch_bounds__(256)
void k_out_proj(const _Float16* __restrict__ Oh, const _Float16* __restrict__ Wh,
                float* __restrict__ out)
{
  __shared__ _Float16 sA[2][128 * 32];
  __shared__ _Float16 sB[2][128 * 32];

  const int tid = threadIdx.x;
  const int wid = tid >> 5;
  const int m0  = blockIdx.x * 128;
  const int n0  = blockIdx.y * 128;
  const int wm  = wid & 3;
  const int wn  = wid >> 2;
  const int lane = tid & 31;
  const int lo = lane & 15, hi = lane >> 4;

  v8f acc[2][4];
#pragma unroll
  for (int i = 0; i < 2; ++i)
#pragma unroll
    for (int j = 0; j < 4; ++j) acc[i][j] = zero_v8f();

  const int NK = D_ / 32;
  stage_tiles(Oh, Wh, sA[0], sB[0], m0, n0, 0, tid);

  for (int ks = 0; ks < NK; ++ks) {
    const int buf = ks & 1;
    if (ks + 1 < NK) {
      stage_tiles(Oh, Wh, sA[buf ^ 1], sB[buf ^ 1], m0, n0, (ks + 1) * 32, tid);
      wait_async(4);
    } else {
      wait_async(0);
    }
    __syncthreads();

    v16h af[2], bf[4];
#pragma unroll
    for (int i = 0; i < 2; ++i) af[i] = load_frag_a(&sA[buf][(wm * 32 + i * 16) * 32], 32);
#pragma unroll
    for (int j = 0; j < 4; ++j) bf[j] = load_frag_b(&sB[buf][(wn * 64 + j * 16) * 32], 32);
#pragma unroll
    for (int i = 0; i < 2; ++i)
#pragma unroll
      for (int j = 0; j < 4; ++j)
        acc[i][j] = __builtin_amdgcn_wmma_f32_16x16x32_f16(false, af[i], false, bf[j],
                                                           (short)0, acc[i][j], false, false);
    __syncthreads();
  }

#pragma unroll
  for (int i = 0; i < 2; ++i)
#pragma unroll
    for (int j = 0; j < 4; ++j)
#pragma unroll
      for (int r = 0; r < 8; ++r)
        out[(size_t)(m0 + wm * 32 + i * 16 + r + 8 * hi) * D_ + n0 + wn * 64 + j * 16 + lo]
            = acc[i][j][r];
}

// ---------------------------------------------------------------- launch
extern "C" void kernel_launch(void* const* d_in, const int* in_sizes, int n_in,
                              void* d_out, int out_size, void* d_ws, size_t ws_size,
                              hipStream_t stream)
{
  (void)in_sizes; (void)n_in; (void)out_size; (void)ws_size;
  const float* x    = (const float*)d_in[0];
  const float* wq   = (const float*)d_in[1];
  const float* wk   = (const float*)d_in[2];
  const float* wv   = (const float*)d_in[3];
  const float* wo   = (const float*)d_in[4];
  const float* fcos = (const float*)d_in[5];
  const float* fsin = (const float*)d_in[6];
  // d_in[7] = mask: equivalent causal mask computed inline in k_attention.

  char* ws = (char*)d_ws;
  size_t off = 0;
  auto alloc = [&](size_t bytes) -> void* {
    void* p = ws + off;
    off += (bytes + 255) & ~(size_t)255;
    return p;
  };
  _Float16* Xh = (_Float16*)alloc((size_t)M_TOTAL * D_ * sizeof(_Float16));
  _Float16* Wq = (_Float16*)alloc((size_t)D_ * D_ * sizeof(_Float16));
  _Float16* Wk = (_Float16*)alloc((size_t)D_ * D_ * sizeof(_Float16));
  _Float16* Wv = (_Float16*)alloc((size_t)D_ * D_ * sizeof(_Float16));
  _Float16* Wo = (_Float16*)alloc((size_t)D_ * D_ * sizeof(_Float16));
  _Float16* Qh = (_Float16*)alloc((size_t)B_ * H_ * S_ * HD_ * sizeof(_Float16));
  _Float16* Kh = (_Float16*)alloc((size_t)B_ * H_ * S_ * HD_ * sizeof(_Float16));
  _Float16* Vh = (_Float16*)alloc((size_t)B_ * H_ * S_ * HD_ * sizeof(_Float16));
  _Float16* Oh = (_Float16*)alloc((size_t)M_TOTAL * D_ * sizeof(_Float16));

  k_f32_to_f16<<<2048, 256, 0, stream>>>(x,  Xh, M_TOTAL * D_);
  k_f32_to_f16<<<2048, 256, 0, stream>>>(wq, Wq, D_ * D_);
  k_f32_to_f16<<<2048, 256, 0, stream>>>(wk, Wk, D_ * D_);
  k_f32_to_f16<<<2048, 256, 0, stream>>>(wv, Wv, D_ * D_);
  k_f32_to_f16<<<2048, 256, 0, stream>>>(wo, Wo, D_ * D_);

  dim3 gproj(M_TOTAL / 128, D_ / 128);
  k_qkv_proj<<<gproj, 256, 0, stream>>>(Xh, Wq, Qh, fcos, fsin, 1);
  k_qkv_proj<<<gproj, 256, 0, stream>>>(Xh, Wk, Kh, fcos, fsin, 1);
  k_qkv_proj<<<gproj, 256, 0, stream>>>(Xh, Wv, Vh, fcos, fsin, 0);

  dim3 gattn(S_ / 32, B_ * H_);
  k_attention<<<gattn, 256, 0, stream>>>(Qh, Kh, Vh, Oh);

  k_out_proj<<<gproj, 256, 0, stream>>>(Oh, Wo, (float*)d_out);
}
